// ViTLinBertSelfAttention_240518168724
// MI455X (gfx1250) — matrix operand order repaired and checked by reading, not compiled, not run on hardware
//
#include <hip/hip_runtime.h>

// ---------------------------------------------------------------------------
// ViT LinBert linear-attention forward, gfx1250 (MI455X), bf16 WMMA.
// B=32, N=1024, D=768, H=12, d=64.
// All WMMA operands are stored so each lane's fragment is contiguous
// 16-byte runs -> global_load_b128, no VALU packing.
//   A operands: row-major        (lane run: two 8-elem bf16 runs)
//   B operands: K-major (transp) (lane run: two 8-elem bf16 runs)
// ---------------------------------------------------------------------------

typedef __attribute__((ext_vector_type(16))) __bf16 v16bf;
typedef __attribute__((ext_vector_type(8)))  __bf16 v8bf;
typedef __attribute__((ext_vector_type(8)))  float  v8f;

union Frag16 {
    v16bf v;
    v8bf  h[2];
    __bf16 e[16];
};

__device__ __forceinline__ __bf16 f2bf(float f) {
    union { float f; unsigned int u; } c; c.f = f;
    unsigned int u = c.u;
    unsigned int r = u + 0x7FFFu + ((u >> 16) & 1u);   // round-to-nearest-even
    union { unsigned short s; __bf16 b; } o; o.s = (unsigned short)(r >> 16);
    return o.b;
}

__device__ __forceinline__ float bf2f(__bf16 b) {
    union { unsigned short s; __bf16 b; } c; c.b = b;
    union { unsigned int u; float f; } o; o.u = ((unsigned int)c.s) << 16;
    return o.f;
}

// A fragment: rows are lane%16; lane's K values = [8*kg, 8*kg+8) u [16+8*kg, 24+8*kg)
__device__ __forceinline__ v16bf frag_a(const __bf16* __restrict__ row, int kg) {
    Frag16 f;
    f.h[0] = *(const v8bf*)(row + 8 * kg);
    f.h[1] = *(const v8bf*)(row + 16 + 8 * kg);
    return f.v;
}
// B fragment from K-major storage: col ptr = BT + n*ldk (+kk); lane's K values = [8*kg, 8*kg+8) u [16+8*kg, 24+8*kg)
__device__ __forceinline__ v16bf frag_b(const __bf16* __restrict__ col, int kg) {
    Frag16 f;
    f.h[0] = *(const v8bf*)(col + 8 * kg);
    f.h[1] = *(const v8bf*)(col + 16 + 8 * kg);
    return f.v;
}

#define WMMA_BF16(A, Bm, C) \
    __builtin_amdgcn_wmma_f32_16x16x32_bf16(false, (A), false, (Bm), (short)0, (C), false, false)

// ---------------------------------------------------------------------------
// Pre-pass kernels: fp32 -> bf16 conversions / weight transposes (K-major)
// ---------------------------------------------------------------------------
__global__ __launch_bounds__(256) void cvt_x_kernel(
    const float* __restrict__ x, __bf16* __restrict__ xbf)
{
    size_t i = (size_t)blockIdx.x * 256 + threadIdx.x;   // 25,165,824 elems
    xbf[i] = f2bf(x[i]);
}

// WT[n*K + k] = W[k*N + n]   (bf16, K-major)
__global__ __launch_bounds__(256) void tr_w_kernel(
    const float* __restrict__ W, __bf16* __restrict__ WT, int K, int Ncols)
{
    size_t i = (size_t)blockIdx.x * 256 + threadIdx.x;   // over K*N
    int n = (int)(i / K), k = (int)(i % K);
    WT[i] = f2bf(W[(size_t)k * Ncols + n]);
}

// ---------------------------------------------------------------------------
// Kernel 1: qkv = x @ W_qkv.  Wave computes 32(M) x 64(N); 8 WMMA / K-step.
// phi applied to q,k.  q stored token-major [bh][tok][64]; phi(k),v stored
// dim-major [bh][64][tok] (K-major for the kv GEMM).
// grid = (1024, 36), block = 32
// ---------------------------------------------------------------------------
__global__ __launch_bounds__(32) void qkv_kernel(
    const __bf16* __restrict__ xbf, const __bf16* __restrict__ WqkvT,
    __bf16* __restrict__ qphi, __bf16* __restrict__ ktb,
    __bf16* __restrict__ vtb)
{
    const int lane = threadIdx.x & 31;
    const int tm = blockIdx.x, ns = blockIdx.y;
    const int kg = lane >> 4, nl = lane & 15;

    const __bf16* arow0 = xbf + (size_t)(tm * 32 + nl) * 768;
    const __bf16* arow1 = arow0 + (size_t)16 * 768;
    const __bf16* bcol  = WqkvT + (size_t)(ns * 64 + nl) * 768;

    v8f acc[2][4] = {};
    for (int kk = 0; kk < 768; kk += 32) {
        v16bf a0 = frag_a(arow0 + kk, kg);
        v16bf a1 = frag_a(arow1 + kk, kg);
#pragma unroll
        for (int t = 0; t < 4; ++t) {
            v16bf b = frag_b(bcol + (size_t)t * 16 * 768 + kk, kg);
            acc[0][t] = WMMA_BF16(a0, b, acc[0][t]);
            acc[1][t] = WMMA_BF16(a1, b, acc[1][t]);
        }
    }

#pragma unroll
    for (int half = 0; half < 2; ++half) {
#pragma unroll
        for (int t = 0; t < 4; ++t) {
            int ccol  = ns * 64 + t * 16 + nl;
            int which = ccol / 768;           // uniform per (block,t)
            int rem   = ccol % 768;
            int h     = rem >> 6;
            int dim   = rem & 63;
#pragma unroll
            for (int r = 0; r < 8; ++r) {
                int gr = tm * 32 + half * 16 + r + 8 * kg;
                int b  = gr >> 10, n = gr & 1023;
                int bh = b * 12 + h;
                float val = acc[half][t][r];
                if (which == 0) {
                    float p = val > 0.0f ? val + 1.0f : __expf(val);
                    qphi[((size_t)bh * 1024 + n) * 64 + dim] = f2bf(p);
                } else if (which == 1) {
                    float p = val > 0.0f ? val + 1.0f : __expf(val);
                    ktb[((size_t)bh * 64 + dim) * 1024 + n] = f2bf(p);
                } else {
                    vtb[((size_t)bh * 64 + dim) * 1024 + n] = f2bf(val);
                }
            }
        }
    }
}

// ---------------------------------------------------------------------------
// Kernel 2: per (b,h) kv[64,64] = phi(K)^T @ V.
// A = Kt (dim-major, row-major [64,1024]); B = Vt (K-major per out column).
// Wave computes a 32x32 quadrant (4 WMMA / K-step). Output stored transposed
// (kvT[n][k]) so the attn GEMM's B operand is K-major.
// grid = 384, block = 128
// ---------------------------------------------------------------------------
__global__ __launch_bounds__(128) void kv_kernel(
    const __bf16* __restrict__ ktb, const __bf16* __restrict__ vtb,
    __bf16* __restrict__ kvT)
{
    const int bh   = blockIdx.x;
    const int wave = threadIdx.x >> 5;
    const int lane = threadIdx.x & 31;
    const int mq = wave >> 1, nq = wave & 1;
    const int kg = lane >> 4, nl = lane & 15;

    const __bf16* KT = ktb + (size_t)bh * 64 * 1024;
    const __bf16* VT = vtb + (size_t)bh * 64 * 1024;

    const __bf16* arow0 = KT + (size_t)(mq * 32 + nl) * 1024;
    const __bf16* arow1 = arow0 + (size_t)16 * 1024;
    const __bf16* bcol0 = VT + (size_t)(nq * 32 + nl) * 1024;
    const __bf16* bcol1 = bcol0 + (size_t)16 * 1024;

    v8f acc[2][2] = {};
    for (int s = 0; s < 1024; s += 32) {
        v16bf a0 = frag_a(arow0 + s, kg);
        v16bf a1 = frag_a(arow1 + s, kg);
        v16bf b0 = frag_b(bcol0 + s, kg);
        v16bf b1 = frag_b(bcol1 + s, kg);
        acc[0][0] = WMMA_BF16(a0, b0, acc[0][0]);
        acc[0][1] = WMMA_BF16(a0, b1, acc[0][1]);
        acc[1][0] = WMMA_BF16(a1, b0, acc[1][0]);
        acc[1][1] = WMMA_BF16(a1, b1, acc[1][1]);
    }
#pragma unroll
    for (int i = 0; i < 2; ++i) {
#pragma unroll
        for (int j = 0; j < 2; ++j) {
#pragma unroll
            for (int r = 0; r < 8; ++r) {
                int m = mq * 32 + i * 16 + r + 8 * kg;   // K-dim index of kv
                int n = nq * 32 + j * 16 + nl;           // out-dim index
                kvT[(size_t)bh * 4096 + (size_t)n * 64 + m] = f2bf(acc[i][j][r]);
            }
        }
    }
}

// ---------------------------------------------------------------------------
// Kernel 3: ksum[bh][d] = 1 + sum_s phi(K)[s][d]  (contiguous in Kt layout)
// ---------------------------------------------------------------------------
__global__ __launch_bounds__(64) void ksum_kernel(
    const __bf16* __restrict__ ktb, float* __restrict__ ksum)
{
    const int bh = blockIdx.x, d = threadIdx.x;
    const __bf16* K = ktb + ((size_t)bh * 64 + d) * 1024;
    float s = 1.0f;                        // phi(0)=1 from the padded token
#pragma unroll 8
    for (int i = 0; i < 1024; ++i) s += bf2f(K[i]);
    ksum[bh * 64 + d] = s;
}

// ---------------------------------------------------------------------------
// Kernel 4: z[bh][n] = 1 / (phi(q)[n] . ksum[bh] + eps)
// ---------------------------------------------------------------------------
__global__ __launch_bounds__(256) void z_kernel(
    const __bf16* __restrict__ qphi, const float* __restrict__ ksum,
    float* __restrict__ zbuf)
{
    const int gid = blockIdx.x * 256 + threadIdx.x;   // bh*1024 + n
    const int bh  = gid >> 10;
    const __bf16* q = qphi + (size_t)gid * 64;
    const float* ks = ksum + bh * 64;
    float acc = 0.0f;
#pragma unroll 8
    for (int i = 0; i < 64; ++i) acc += bf2f(q[i]) * ks[i];
    zbuf[gid] = 1.0f / (acc + 1e-6f);
}

// ---------------------------------------------------------------------------
// Kernel 5: attn = (phi(Q) @ kv) * z, scattered to [B,N,768] bf16 (row-major).
// A = qphi (row-major), B = kvT (K-major). Wave: 32 tokens x 64 dims.
// grid = (384, 8), block = 128
// ---------------------------------------------------------------------------
__global__ __launch_bounds__(128) void attn_kernel(
    const __bf16* __restrict__ qphi, const __bf16* __restrict__ kvT,
    const float* __restrict__ zbuf, __bf16* __restrict__ attn)
{
    const int bh   = blockIdx.x;
    const int wave = threadIdx.x >> 5;
    const int lane = threadIdx.x & 31;
    const int mt = blockIdx.y * 4 + wave;    // 32-token strip, 0..31
    const int b  = bh / 12, h = bh % 12;
    const int kg = lane >> 4, nl = lane & 15;

    const __bf16* Q  = qphi + (size_t)bh * 1024 * 64;
    const __bf16* KV = kvT + (size_t)bh * 4096;

    const __bf16* arow0 = Q + (size_t)(mt * 32 + nl) * 64;
    const __bf16* arow1 = arow0 + (size_t)16 * 64;

    v8f acc[2][4] = {};
#pragma unroll
    for (int kk = 0; kk < 64; kk += 32) {
        v16bf a0 = frag_a(arow0 + kk, kg);
        v16bf a1 = frag_a(arow1 + kk, kg);
#pragma unroll
        for (int t = 0; t < 4; ++t) {
            v16bf bm = frag_b(KV + (size_t)(t * 16 + nl) * 64 + kk, kg);
            acc[0][t] = WMMA_BF16(a0, bm, acc[0][t]);
            acc[1][t] = WMMA_BF16(a1, bm, acc[1][t]);
        }
    }
#pragma unroll
    for (int half = 0; half < 2; ++half) {
#pragma unroll
        for (int r = 0; r < 8; ++r) {
            int tok = mt * 32 + half * 16 + r + 8 * kg;
            float zz = zbuf[bh * 1024 + tok];
            __bf16* dst = attn + ((size_t)(b * 1024 + tok)) * 768 + h * 64;
#pragma unroll
            for (int t = 0; t < 4; ++t)
                dst[t * 16 + nl] = f2bf(acc[half][t][r] * zz);
        }
    }
}

// ---------------------------------------------------------------------------
// Kernel 6: out = attn @ W_out + b_out (fp32 out). Wave: 32x64, 8 WMMA/K-step.
// grid = (1024, 12), block = 32
// ---------------------------------------------------------------------------
__global__ __launch_bounds__(32) void outproj_kernel(
    const __bf16* __restrict__ attn, const __bf16* __restrict__ WoutT,
    const float* __restrict__ bout, float* __restrict__ out)
{
    const int lane = threadIdx.x & 31;
    const int tm = blockIdx.x, ns = blockIdx.y;
    const int kg = lane >> 4, nl = lane & 15;

    const __bf16* arow0 = attn + (size_t)(tm * 32 + nl) * 768;
    const __bf16* arow1 = arow0 + (size_t)16 * 768;
    const __bf16* bcol  = WoutT + (size_t)(ns * 64 + nl) * 768;

    v8f acc[2][4] = {};
    for (int kk = 0; kk < 768; kk += 32) {
        v16bf a0 = frag_a(arow0 + kk, kg);
        v16bf a1 = frag_a(arow1 + kk, kg);
#pragma unroll
        for (int t = 0; t < 4; ++t) {
            v16bf b = frag_b(bcol + (size_t)t * 16 * 768 + kk, kg);
            acc[0][t] = WMMA_BF16(a0, b, acc[0][t]);
            acc[1][t] = WMMA_BF16(a1, b, acc[1][t]);
        }
    }
#pragma unroll
    for (int t = 0; t < 4; ++t) {
        int c = ns * 64 + t * 16 + nl;
        float bias = bout[c];
#pragma unroll
        for (int half = 0; half < 2; ++half) {
#pragma unroll
            for (int r = 0; r < 8; ++r) {
                int gr = tm * 32 + half * 16 + r + 8 * kg;
                out[(size_t)gr * 768 + c] = acc[half][t][r] + bias;
            }
        }
    }
}

// ---------------------------------------------------------------------------
extern "C" void kernel_launch(void* const* d_in, const int* in_sizes, int n_in,
                              void* d_out, int out_size, void* d_ws, size_t ws_size,
                              hipStream_t stream)
{
    const float* x    = (const float*)d_in[0];   // [32,1024,768]
    const float* Wqkv = (const float*)d_in[1];   // [768,2304]
    const float* Wout = (const float*)d_in[2];   // [768,768]
    const float* bout = (const float*)d_in[3];   // [768]
    float* out = (float*)d_out;                  // [32,1024,768]

    char* ws = (char*)d_ws;
    const size_t SZ_TOK = (size_t)32 * 1024 * 768 * 2;        // 50,331,648 B
    size_t off = 0;
    __bf16* xbf   = (__bf16*)(ws + off); off += SZ_TOK;       // x as bf16
    __bf16* qphi  = (__bf16*)(ws + off); off += SZ_TOK;       // [bh][tok][64]
    __bf16* ktb   = (__bf16*)(ws + off); off += SZ_TOK;       // [bh][64][tok]
    __bf16* vtb   = (__bf16*)(ws + off); off += SZ_TOK;       // [bh][64][tok]
    __bf16* attn  = (__bf16*)(ws + off); off += SZ_TOK;       // [B*N][768]
    __bf16* kvT   = (__bf16*)(ws + off); off += (size_t)384 * 4096 * 2;
    __bf16* WqkvT = (__bf16*)(ws + off); off += (size_t)2304 * 768 * 2;
    __bf16* WoutT = (__bf16*)(ws + off); off += (size_t)768 * 768 * 2;
    float*  ksum  = (float*)(ws + off);  off += (size_t)384 * 64 * 4;
    float*  zbuf  = (float*)(ws + off);  off += (size_t)384 * 1024 * 4;

    cvt_x_kernel  <<<dim3(98304),    256, 0, stream>>>(x, xbf);
    tr_w_kernel   <<<dim3(6912),     256, 0, stream>>>(Wqkv, WqkvT, 768, 2304);
    tr_w_kernel   <<<dim3(2304),     256, 0, stream>>>(Wout, WoutT, 768, 768);

    qkv_kernel    <<<dim3(1024, 36), 32,  0, stream>>>(xbf, WqkvT, qphi, ktb, vtb);
    kv_kernel     <<<dim3(384),      128, 0, stream>>>(ktb, vtb, kvT);
    ksum_kernel   <<<dim3(384),      64,  0, stream>>>(ktb, ksum);
    z_kernel      <<<dim3(384 * 4),  256, 0, stream>>>(qphi, ksum, zbuf);
    attn_kernel   <<<dim3(384, 8),   128, 0, stream>>>(qphi, kvT, zbuf, attn);
    outproj_kernel<<<dim3(1024, 12), 32,  0, stream>>>(attn, WoutT, bout, out);
}
